// RAM_22789096473275
// MI455X (gfx1250) — matrix operand order repaired and checked by reading, not compiled, not run on hardware
//
#include <hip/hip_runtime.h>
#include <math.h>

// ---------------- problem constants ----------------
constexpr int B = 256, S = 256, E = 300, H = 300, A = 16, POL = 3, LAYERS = 4;
constexpr int KP  = 320;   // padded E (and padded H) for K dimension
constexpr int KC  = 640;   // combined K: [x(320) | h(320)]
constexpr int NROW = 1216; // 4H=1200 padded to cover unaligned gate tiles
constexpr int MDS = 608;   // mem row stride (601 used)
constexpr int DL  = 601;   // 2H+1

typedef __attribute__((ext_vector_type(16))) __bf16 v16bf;
typedef __attribute__((ext_vector_type(8)))  float  v8f;
union ABf { uint4 q[2]; v16bf v; };

__device__ __forceinline__ unsigned short f2bf(float f) {
  unsigned int u = __float_as_uint(f);
  u += 0x7fffu + ((u >> 16) & 1u);            // round-to-nearest-even
  return (unsigned short)(u >> 16);
}
__device__ __forceinline__ float sigf(float x) { return 1.f / (1.f + __expf(-x)); }

// ----- CDNA5 async global->LDS copy (16B per lane), tracked by ASYNCcnt -----
__device__ __forceinline__ void async_ld_b128(const unsigned short* __restrict__ src,
                                              unsigned short* dst)
{
  // VDST = VGPR with LDS byte address; VADDR = 64-bit global address.
  asm volatile("global_load_async_to_lds_b128 %0, %1, off"
               :: "v"((unsigned)(size_t)dst), "v"(src)
               : "memory");
}
__device__ __forceinline__ void wait_async0() {
#if __has_builtin(__builtin_amdgcn_s_wait_asynccnt)
  __builtin_amdgcn_s_wait_asynccnt(0);
#else
  asm volatile("s_wait_asynccnt 0" ::: "memory");
#endif
}

// ---------------- prep: lengths + aspect mean embedding ----------------
__global__ __launch_bounds__(256) void k_prep(
    const int* __restrict__ text, const int* __restrict__ aspect,
    const int* __restrict__ left, const float* __restrict__ embed,
    float* __restrict__ asp, float* __restrict__ lens)
{
  const int b = blockIdx.x, tid = threadIdx.x;
  __shared__ float red[256];
  __shared__ float s_alen;

  red[tid] = (text[b * S + tid] != 0) ? 1.f : 0.f; __syncthreads();
  for (int st = 128; st > 0; st >>= 1) { if (tid < st) red[tid] += red[tid + st]; __syncthreads(); }
  if (tid == 0) lens[B + b] = red[0];
  __syncthreads();

  red[tid] = (left[b * S + tid] != 0) ? 1.f : 0.f; __syncthreads();
  for (int st = 128; st > 0; st >>= 1) { if (tid < st) red[tid] += red[tid + st]; __syncthreads(); }
  if (tid == 0) lens[b] = red[0];
  __syncthreads();

  red[tid] = (tid < A && aspect[b * A + tid] != 0) ? 1.f : 0.f; __syncthreads();
  for (int st = 128; st > 0; st >>= 1) { if (tid < st) red[tid] += red[tid + st]; __syncthreads(); }
  if (tid == 0) { lens[2 * B + b] = red[0]; s_alen = red[0]; }
  __syncthreads();

  for (int e = tid; e < E; e += 256) {
    float acc = 0.f;
    for (int a = 0; a < A; ++a) acc += embed[(size_t)aspect[b * A + a] * E + e];
    asp[b * E + e] = acc / s_alen;
  }
}

// ---------------- embedding gather -> bf16 (K-padded) ----------------
__global__ __launch_bounds__(KP) void k_embed(
    const int* __restrict__ text, const float* __restrict__ embed,
    unsigned short* __restrict__ xbf)
{
  const size_t idx = blockIdx.x;              // b*S + s
  const int tok = text[idx];
  const int k = threadIdx.x;                  // 0..319
  unsigned short o = 0;
  if (k < E) o = f2bf(embed[(size_t)tok * E + k]);
  xbf[idx * KP + k] = o;
}

// ---------------- combined [Wih | Whh] bf16 weights, row-padded ----------------
__global__ __launch_bounds__(KC) void k_wc(
    const float* __restrict__ wih, const float* __restrict__ whh,
    unsigned short* __restrict__ wc)
{
  const int n = blockIdx.x;                   // 0..NROW-1
  const int k = threadIdx.x;                  // 0..639
  unsigned short o = 0;
  if (n < 4 * H) {
    if (k < KP) { if (k < E) o = f2bf(wih[(size_t)n * E + k]); }
    else        { int kk = k - KP; if (kk < H) o = f2bf(whh[(size_t)n * H + kk]); }
  }
  wc[(size_t)n * KC + k] = o;
}

// ---------------- one WMMA K-chunk: A(16x32 bf16) x 4 gate tiles ----------------
__device__ __forceinline__ void wstep(
    const unsigned short (*am)[KP], int k0,
    const unsigned short* __restrict__ wc, int wkoff, int j0, int lane,
    v8f& a0, v8f& a1, v8f& a2, v8f& a3)
{
  ABf Am, Bm;
  const int m  = lane & 15;
  const int ka = k0 + ((lane & 16) ? 8 : 0);
  Am.q[0] = *(const uint4*)&am[m][ka];
  Am.q[1] = *(const uint4*)&am[m][ka + 16];
  const int kb = wkoff + k0 + ((lane & 16) ? 8 : 0);
  const int nr = j0 + (lane & 15);
#pragma unroll
  for (int g = 0; g < 4; ++g) {
    const unsigned short* wr = wc + (size_t)(g * H + nr) * KC + kb;
    Bm.q[0] = *(const uint4*)wr;
    Bm.q[1] = *(const uint4*)(wr + 16);
    v8f& acc = (g == 0) ? a0 : (g == 1) ? a1 : (g == 2) ? a2 : a3;
    acc = __builtin_amdgcn_wmma_f32_16x16x32_bf16(false, Am.v, false, Bm.v,
                                                  (short)0, acc, false, false);
  }
}

// ---------------- BiLSTM: persistent per batch-slice, WMMA bf16 ----------------
__global__ __launch_bounds__(256) void k_lstm(
    const unsigned short* __restrict__ xbf,
    const unsigned short* __restrict__ wcf,
    const unsigned short* __restrict__ wcb,
    const float* __restrict__ fbv, const float* __restrict__ bbv,
    float* __restrict__ mem)
{
  const int tid  = threadIdx.x, lane = tid & 31, wid = tid >> 5;
  const int bo   = blockIdx.x * 16;
  const int dir  = blockIdx.y;
  const unsigned short* __restrict__ wc   = dir ? wcb : wcf;
  const float* __restrict__ bias          = dir ? bbv : fbv;

  __shared__ unsigned short xsm[2][16][KP];   // double-buffered x_t slice
  __shared__ unsigned short hsm[2][16][KP];   // double-buffered h slice

  // zero both h buffers (h0 = 0, pad cols stay 0 forever)
  for (int u = tid; u < 2 * 16 * (KP / 8); u += 256)
    ((uint4*)&hsm[0][0][0])[u] = make_uint4(0, 0, 0, 0);
  // stage x for step 0 via async DMA to LDS
  {
    const int s0 = dir ? (S - 1) : 0;
    for (int u = tid; u < 16 * (KP / 8); u += 256) {
      const int r = u / (KP / 8), c = u % (KP / 8);
      async_ld_b128(&xbf[((size_t)(bo + r) * S + s0) * KP + c * 8],
                    &xsm[0][r][c * 8]);
    }
  }

  float cst[3][8];       // cell state, lives in registers in WMMA C/D layout
  float brg[3][4];       // per-lane gate biases
#pragma unroll
  for (int ti = 0; ti < 3; ++ti) {
#pragma unroll
    for (int v = 0; v < 8; ++v) cst[ti][v] = 0.f;
    const int jt = wid + 8 * ti;
    const int j  = jt * 16 + (lane & 15);
#pragma unroll
    for (int g = 0; g < 4; ++g)
      brg[ti][g] = (jt < 19 && j < H) ? bias[g * H + j] : 0.f;
  }
  wait_async0();
  __syncthreads();

  int p = 0;
  for (int st = 0; st < S; ++st) {
    const int s = dir ? (S - 1 - st) : st;
    const unsigned short (*xr)[KP] = xsm[p];
    const unsigned short (*hr)[KP] = hsm[p];
    unsigned short (*hw)[KP]       = hsm[p ^ 1];

    // kick off async staging of next step's x slice first: the DMA overlaps
    // the entire GEMM + gate phase and is only drained at the barrier.
    if (st + 1 < S) {
      const int s2 = dir ? (S - 2 - st) : (st + 1);
      for (int u = tid; u < 16 * (KP / 8); u += 256) {
        const int r = u / (KP / 8), c = u % (KP / 8);
        async_ld_b128(&xbf[((size_t)(bo + r) * S + s2) * KP + c * 8],
                      &xsm[p ^ 1][r][c * 8]);
      }
      if (st + 2 < S) {
        const int s3 = dir ? (S - 3 - st) : (st + 2);
        __builtin_prefetch(&xbf[((size_t)(bo + (tid & 15)) * S + s3) * KP], 0, 0);
      }
    }

#pragma unroll
    for (int ti = 0; ti < 3; ++ti) {
      const int jt = wid + 8 * ti;
      if (jt >= 19) break;                       // wave-uniform: EXEC stays full
      const int j0 = jt * 16;
      v8f a0 = {0,0,0,0,0,0,0,0}, a1 = a0, a2 = a0, a3 = a0;
#pragma unroll 4
      for (int k0 = 0; k0 < KP; k0 += 32)        // x @ Wih^T
        wstep(xr, k0, wc, 0, j0, lane, a0, a1, a2, a3);
#pragma unroll 4
      for (int k0 = 0; k0 < KP; k0 += 32)        // h @ Whh^T
        wstep(hr, k0, wc, KP, j0, lane, a0, a1, a2, a3);

      // gate nonlinearity fully in registers (C/D layout: m = v + 8*(lane>=16))
      const int  j  = j0 + (lane & 15);
      const bool jv = (j < H);
      const int  mb = (lane >> 4) << 3;
#pragma unroll
      for (int v = 0; v < 8; ++v) {
        const float ig = a0[v] + brg[ti][0];
        const float fg = a1[v] + brg[ti][1];
        const float gg = a2[v] + brg[ti][2];
        const float og = a3[v] + brg[ti][3];
        const float c  = sigf(fg) * cst[ti][v] + sigf(ig) * tanhf(gg);
        cst[ti][v] = c;
        const float h = sigf(og) * tanhf(c);
        if (jv) {
          const int m = mb + v;
          hw[m][j] = f2bf(h);
          mem[(((size_t)(bo + m)) * S + s) * MDS + dir * H + j] = h;
        }
      }
    }

    // drain this wave's async DMA, then make it workgroup-visible.
    wait_async0();
    __syncthreads();
    p ^= 1;
  }
}

// ---------------- location weighting (in place) ----------------
__global__ __launch_bounds__(128) void k_loc(float* __restrict__ mem,
                                             const float* __restrict__ lens)
{
  const int s = blockIdx.x, b = blockIdx.y, tid = threadIdx.x;
  const float ll = lens[b], Ma = lens[B + b], al = lens[2 * B + b];
  const float La = ll + al, fs = (float)s;
  float w, u;
  if      (fs < ll) { w = 1.f - (ll - fs) / Ma;       u = fs - ll; }
  else if (fs < La) { w = 1.f;                        u = 0.f; }
  else if (fs < Ma) { w = 1.f - (fs - La + 1.f) / Ma; u = fs - La + 1.f; }
  else              { w = 0.f;                        u = 1.f; }
  float* row = mem + ((size_t)(b * S + s)) * MDS;
  for (int d = tid; d < 2 * H; d += 128) row[d] *= w;
  if (tid == 0) row[2 * H] = u;
}

__global__ void k_zero(float* __restrict__ p, int n) {
  int i = blockIdx.x * blockDim.x + threadIdx.x;
  if (i < n) p[i] = 0.f;
}

// ---------------- attention: per-batch constant part of the score ----------------
__global__ __launch_bounds__(256) void k_pb(
    const float* __restrict__ et, const float* __restrict__ asp,
    const float* __restrict__ aw, const float* __restrict__ ab,
    float* __restrict__ pb)
{
  const int b = blockIdx.x, tid = threadIdx.x;
  __shared__ float red[256];
  float acc = 0.f;
  for (int k = tid; k < E; k += 256)
    acc += et[b * E + k] * aw[DL + k] + asp[b * E + k] * aw[DL + E + k];
  red[tid] = acc; __syncthreads();
  for (int st = 128; st > 0; st >>= 1) { if (tid < st) red[tid] += red[tid + st]; __syncthreads(); }
  if (tid == 0) pb[b] = red[0] + ab[0];
}

// ---------------- attention: score per (b,s) ----------------
__global__ __launch_bounds__(64) void k_gdot(
    const float* __restrict__ mem, const float* __restrict__ aw,
    const float* __restrict__ pb, float* __restrict__ g)
{
  const int s = blockIdx.x, b = blockIdx.y, tid = threadIdx.x;
  const float* row = mem + ((size_t)(b * S + s)) * MDS;
  float acc = 0.f;
  for (int d = tid; d < DL; d += 64) acc += row[d] * aw[d];
  __shared__ float red[64];
  red[tid] = acc; __syncthreads();
  for (int st = 32; st > 0; st >>= 1) { if (tid < st) red[tid] += red[tid + st]; __syncthreads(); }
  if (tid == 0) g[s * B + b] = red[0] + pb[b];
}

// ---------------- softmax over batch axis (faithful: dim=0) ----------------
__global__ __launch_bounds__(B) void k_soft(const float* __restrict__ g,
                                            float* __restrict__ alpha)
{
  const int s = blockIdx.x, b = threadIdx.x;
  __shared__ float red[B];
  const float v = g[s * B + b];
  red[b] = v; __syncthreads();
  for (int st = 128; st > 0; st >>= 1) { if (b < st) red[b] = fmaxf(red[b], red[b + st]); __syncthreads(); }
  const float mx = red[0]; __syncthreads();
  const float e = __expf(v - mx);
  red[b] = e; __syncthreads();
  for (int st = 128; st > 0; st >>= 1) { if (b < st) red[b] += red[b + st]; __syncthreads(); }
  alpha[b * S + s] = e / red[0];
}

// ---------------- i[b,:] = alpha[b,:] @ mem[b,:,:] ----------------
__global__ __launch_bounds__(640) void k_ivec(
    const float* __restrict__ mem, const float* __restrict__ alpha,
    float* __restrict__ iv)
{
  const int b = blockIdx.x, tid = threadIdx.x;
  __shared__ float al[S];
  if (tid < S) al[tid] = alpha[b * S + tid];
  __syncthreads();
  if (tid < DL) {
    float acc = 0.f;
    const float* base = mem + ((size_t)b * S) * MDS + tid;
    for (int s = 0; s < S; ++s) acc += al[s] * base[(size_t)s * MDS];
    iv[b * DL + tid] = acc;
  }
}

// ---------------- GRU cell: both input/hidden projections ----------------
__global__ __launch_bounds__(900) void k_gru(
    const float* __restrict__ iv, const float* __restrict__ ein,
    const float* __restrict__ wih, const float* __restrict__ whh,
    const float* __restrict__ bih, const float* __restrict__ bhh,
    float* __restrict__ gi, float* __restrict__ gh)
{
  const int b = blockIdx.x, tid = threadIdx.x;
  __shared__ float ivs[DL];
  __shared__ float ets[E];
  for (int k = tid; k < DL; k += 900) ivs[k] = iv[b * DL + k];
  for (int k = tid; k < E;  k += 900) ets[k] = ein[b * E + k];
  __syncthreads();
  const int n = tid;  // < 900 by blockDim
  float a1 = bih[n];
  const float* w1 = wih + (size_t)n * DL;
  for (int k = 0; k < DL; ++k) a1 += ivs[k] * w1[k];
  float a2 = bhh[n];
  const float* w2 = whh + (size_t)n * E;
  for (int k = 0; k < E; ++k) a2 += ets[k] * w2[k];
  gi[b * 900 + n] = a1;
  gh[b * 900 + n] = a2;
}

__global__ __launch_bounds__(320) void k_gcomb(
    const float* __restrict__ gi, const float* __restrict__ gh,
    const float* __restrict__ ein, float* __restrict__ eout)
{
  const int b = blockIdx.x, j = threadIdx.x;
  if (j >= E) return;
  const float r  = sigf(gi[b * 900 + j]       + gh[b * 900 + j]);
  const float z  = sigf(gi[b * 900 + E + j]   + gh[b * 900 + E + j]);
  const float nt = tanhf(gi[b * 900 + 2*E + j] + r * gh[b * 900 + 2*E + j]);
  eout[b * E + j] = (1.f - z) * nt + z * ein[b * E + j];
}

// ---------------- dense head ----------------
__global__ __launch_bounds__(128) void k_dense(
    const float* __restrict__ et, const float* __restrict__ dw,
    const float* __restrict__ db, float* __restrict__ out)
{
  const int idx = blockIdx.x * blockDim.x + threadIdx.x;
  if (idx >= B * POL) return;
  const int b = idx / POL, p = idx % POL;
  float acc = db[p];
  for (int k = 0; k < E; ++k) acc += et[b * E + k] * dw[p * E + k];
  out[idx] = acc;
}

// ---------------- launcher ----------------
extern "C" void kernel_launch(void* const* d_in, const int* in_sizes, int n_in,
                              void* d_out, int out_size, void* d_ws, size_t ws_size,
                              hipStream_t stream)
{
  const int*   text   = (const int*)d_in[0];
  const int*   aspect = (const int*)d_in[1];
  const int*   left   = (const int*)d_in[2];
  const float* embed  = (const float*)d_in[3];
  const float* fw_ih  = (const float*)d_in[4];
  const float* fw_hh  = (const float*)d_in[5];
  const float* fbv    = (const float*)d_in[6];
  const float* bw_ih  = (const float*)d_in[7];
  const float* bw_hh  = (const float*)d_in[8];
  const float* bbv    = (const float*)d_in[9];
  const float* attn_w = (const float*)d_in[10];
  const float* attn_b = (const float*)d_in[11];
  const float* gwih   = (const float*)d_in[12];
  const float* gwhh   = (const float*)d_in[13];
  const float* gbih   = (const float*)d_in[14];
  const float* gbhh   = (const float*)d_in[15];
  const float* dw     = (const float*)d_in[16];
  const float* db     = (const float*)d_in[17];
  float* out = (float*)d_out;

  char* base = (char*)d_ws;
  size_t off = 0;
  auto alloc = [&](size_t bytes) -> void* {
    void* p = base + off;
    off = (off + bytes + 255) & ~(size_t)255;
    return p;
  };
  unsigned short* xbf = (unsigned short*)alloc((size_t)B * S * KP * 2);
  unsigned short* wcf = (unsigned short*)alloc((size_t)NROW * KC * 2);
  unsigned short* wcb = (unsigned short*)alloc((size_t)NROW * KC * 2);
  float* mem  = (float*)alloc((size_t)B * S * MDS * 4);
  float* asp  = (float*)alloc((size_t)B * E * 4);
  float* lens = (float*)alloc((size_t)3 * B * 4);
  float* pb   = (float*)alloc((size_t)B * 4);
  float* gbuf = (float*)alloc((size_t)S * B * 4);
  float* alp  = (float*)alloc((size_t)B * S * 4);
  float* iv   = (float*)alloc((size_t)B * DL * 4);
  float* gi   = (float*)alloc((size_t)B * 900 * 4);
  float* gh   = (float*)alloc((size_t)B * 900 * 4);
  float* et0  = (float*)alloc((size_t)B * E * 4);
  float* et1  = (float*)alloc((size_t)B * E * 4);

  k_prep <<<dim3(B),        dim3(256), 0, stream>>>(text, aspect, left, embed, asp, lens);
  k_embed<<<dim3(B * S),    dim3(KP),  0, stream>>>(text, embed, xbf);
  k_wc   <<<dim3(NROW),     dim3(KC),  0, stream>>>(fw_ih, fw_hh, wcf);
  k_wc   <<<dim3(NROW),     dim3(KC),  0, stream>>>(bw_ih, bw_hh, wcb);
  k_lstm <<<dim3(B / 16, 2), dim3(256), 0, stream>>>(xbf, wcf, wcb, fbv, bbv, mem);
  k_loc  <<<dim3(S, B),     dim3(128), 0, stream>>>(mem, lens);
  k_zero <<<dim3((B * E + 255) / 256), dim3(256), 0, stream>>>(et0, B * E);

  float* ebuf[2] = {et0, et1};
  for (int l = 0; l < LAYERS; ++l) {
    float* ein  = ebuf[l & 1];
    float* eout = ebuf[(l + 1) & 1];
    k_pb   <<<dim3(B),    dim3(256), 0, stream>>>(ein, asp, attn_w, attn_b, pb);
    k_gdot <<<dim3(S, B), dim3(64),  0, stream>>>(mem, attn_w, pb, gbuf);
    k_soft <<<dim3(S),    dim3(B),   0, stream>>>(gbuf, alp);
    k_ivec <<<dim3(B),    dim3(640), 0, stream>>>(mem, alp, iv);
    k_gru  <<<dim3(B),    dim3(900), 0, stream>>>(iv, ein, gwih, gwhh, gbih, gbhh, gi, gh);
    k_gcomb<<<dim3(B),    dim3(320), 0, stream>>>(gi, gh, ein, eout);
  }
  k_dense<<<dim3((B * POL + 127) / 128), dim3(128), 0, stream>>>(ebuf[LAYERS & 1], dw, db, out);
}
